// MoE_86406152061397
// MI455X (gfx1250) — matrix operand order repaired and checked by reading, not compiled, not run on hardware
//
#include <hip/hip_runtime.h>
#include <math.h>

// ---------------------------------------------------------------------------
// MoE top-2 routed FFN for gfx1250 (wave32, WMMA bf16 -> f32 accumulate)
// N=2048 tokens, D=1024, E=16 experts, H=2048, O=1024, TOP_K=2
// GEMM tiling v2: block 256 thr (8 waves), tile 64M x 128N x 64K,
// wave = 16M x 64N (4 accumulators), 8 WMMAs per barrier pair.
// ---------------------------------------------------------------------------

#define N_TOK 2048
#define DIM_D 1024
#define N_EXP 16
#define DIM_H 2048
#define DIM_O 1024
#define TOPK  2

#define BM 64
#define BN 128
#define BK 64
#define BROW 72   // LDS row stride in bf16 elements (144B): conflict-free b128

typedef __attribute__((ext_vector_type(16))) __bf16 v16bf;
typedef __attribute__((ext_vector_type(8)))  float  v8f;

struct U4x2 { uint4 a, b; };

static __device__ __forceinline__ v16bf bitcast_v16bf(uint4 x, uint4 y) {
    U4x2 t{x, y};
    return __builtin_bit_cast(v16bf, t);
}

// ---------------------------------------------------------------------------
// Kernel 0: zero output + counts
// ---------------------------------------------------------------------------
__global__ __launch_bounds__(256)
void moe_zero_kernel(float* __restrict__ out, int total, int* __restrict__ counts) {
    int i = blockIdx.x * blockDim.x + threadIdx.x;
    if (i < total) out[i] = 0.0f;
    if (i < N_EXP) counts[i] = 0;
}

// ---------------------------------------------------------------------------
// Kernel 1: router. One wave32 per token. Lane e (0..15) owns expert e's
// logit; lanes 16..31 compute the other half of D and fold in via shfl_xor.
// ---------------------------------------------------------------------------
__global__ __launch_bounds__(256)
void moe_router_kernel(const float* __restrict__ z, const float* __restrict__ rw,
                       const float* __restrict__ rb,
                       int* __restrict__ topk_idx, float* __restrict__ topk_w,
                       int* __restrict__ counts) {
    const int wid   = threadIdx.x >> 5;
    const int lane  = threadIdx.x & 31;
    const int token = blockIdx.x * 8 + wid;          // grid.x = N/8
    const int e     = lane & 15;
    const int half  = lane >> 4;

    const float* zr = z  + (size_t)token * DIM_D + half * (DIM_D / 2);
    const float* wr = rw + (size_t)e     * DIM_D + half * (DIM_D / 2);
    float acc = 0.0f;
    #pragma unroll 4
    for (int d = 0; d < DIM_D / 2; d += 4) {
        float4 a = *(const float4*)(zr + d);
        float4 b = *(const float4*)(wr + d);
        acc += a.x * b.x + a.y * b.y + a.z * b.z + a.w * b.w;
    }
    acc += __shfl_xor(acc, 16, 32);                  // fold halves
    float logit = acc + rb[e];

    // softmax over the 16-lane group (replicated in both halves)
    float mx = logit;
    for (int m = 8; m >= 1; m >>= 1) mx = fmaxf(mx, __shfl_xor(mx, m, 32));
    float p = expf(logit - mx);
    float s = p;
    for (int m = 8; m >= 1; m >>= 1) s += __shfl_xor(s, m, 32);
    float prob = p / s;

    // top-1 (lower index wins ties, matching jax.lax.top_k)
    float v1 = prob; int i1 = e;
    for (int m = 8; m >= 1; m >>= 1) {
        float ov = __shfl_xor(v1, m, 32); int oi = __shfl_xor(i1, m, 32);
        if (ov > v1 || (ov == v1 && oi < i1)) { v1 = ov; i1 = oi; }
    }
    // top-2
    float v2 = (e == i1) ? -1.0f : prob; int i2 = e;
    for (int m = 8; m >= 1; m >>= 1) {
        float ov = __shfl_xor(v2, m, 32); int oi = __shfl_xor(i2, m, 32);
        if (ov > v2 || (ov == v2 && oi < i2)) { v2 = ov; i2 = oi; }
    }

    if (lane == 0) {
        float s2 = v1 + v2;
        topk_idx[token * 2 + 0] = i1;
        topk_idx[token * 2 + 1] = i2;
        topk_w[token * 2 + 0]   = v1 / s2;
        topk_w[token * 2 + 1]   = v2 / s2;
        atomicAdd(&counts[i1], 1);
        atomicAdd(&counts[i2], 1);
    }
}

// ---------------------------------------------------------------------------
// Kernel 2: tiny exclusive scan over 16 expert counts
// ---------------------------------------------------------------------------
__global__ void moe_scan_kernel(const int* __restrict__ counts,
                                int* __restrict__ segOff, int* __restrict__ cursor) {
    if (threadIdx.x == 0) {
        int off = 0;
        for (int e = 0; e < N_EXP; e++) {
            segOff[e] = off; cursor[e] = off; off += counts[e];
        }
        segOff[N_EXP] = off;   // == N_TOK * TOPK
    }
}

// ---------------------------------------------------------------------------
// Kernel 3: compact (token,slot) assignments into per-expert segments
// ---------------------------------------------------------------------------
__global__ __launch_bounds__(256)
void moe_place_kernel(const int* __restrict__ topk_idx, int* __restrict__ cursor,
                      int* __restrict__ assign) {
    int i = blockIdx.x * blockDim.x + threadIdx.x;   // i = token*2 + slot
    if (i < N_TOK * TOPK) {
        int e = topk_idx[i];
        int pos = atomicAdd(&cursor[e], 1);
        assign[pos] = i;
    }
}

// ---------------------------------------------------------------------------
// WMMA fragment helpers (documented gfx1250 layouts)
//
// A (16x32, 16-bit): lane l holds row m = l&15; halves j=0..7 -> K=(l>>4)*8+j,
//                    j=8..15 -> K = 16 + (l>>4)*8 + (j-8).
// B (32x16, 16-bit): lane l holds col n = l&15; same K mapping as A.
// C/D (16x16 f32):   VGPR r, lane l -> m = r + 8*(l>>4), n = l&15.
// ---------------------------------------------------------------------------
static __device__ __forceinline__ v16bf load_b_frag(const __bf16* Bs, int n_l, int kb) {
    const uint4* p = (const uint4*)(Bs + n_l * BROW + kb);
    return bitcast_v16bf(p[0], p[2]);
}

static __device__ __forceinline__ v16bf load_a_f32(const float* ap) {
    // ap points at row + k0 + half*8 ; two 8-float runs at +0 and +16
    float4 f0 = ((const float4*)ap)[0];
    float4 f1 = ((const float4*)ap)[1];
    float4 f2 = ((const float4*)(ap + 16))[0];
    float4 f3 = ((const float4*)(ap + 16))[1];
    v16bf a;
    a[0]=(__bf16)f0.x; a[1]=(__bf16)f0.y; a[2]=(__bf16)f0.z; a[3]=(__bf16)f0.w;
    a[4]=(__bf16)f1.x; a[5]=(__bf16)f1.y; a[6]=(__bf16)f1.z; a[7]=(__bf16)f1.w;
    a[8]=(__bf16)f2.x; a[9]=(__bf16)f2.y; a[10]=(__bf16)f2.z; a[11]=(__bf16)f2.w;
    a[12]=(__bf16)f3.x; a[13]=(__bf16)f3.y; a[14]=(__bf16)f3.z; a[15]=(__bf16)f3.w;
    return a;
}

static __device__ __forceinline__ v16bf zero_a() {
    v16bf a;
    #pragma unroll
    for (int j = 0; j < 16; j++) a[j] = (__bf16)0.0f;
    return a;
}

// Cooperative B tile load: BN(128) rows x BK(64) K, fp32 -> bf16 into LDS.
// thread t: row = t>>1, kp = (t&1)*32 ; 32 consecutive floats per thread.
static __device__ __forceinline__ void load_b_tile(const float* __restrict__ wrow_base,
                                                   int ldK, int k0, __bf16* Bs,
                                                   int tid, bool prefetch_next) {
    const int n_l = tid >> 1;
    const int kp  = (tid & 1) * 32;
    const float* gp = wrow_base + (size_t)n_l * ldK + k0 + kp;
    __bf16* sp = &Bs[n_l * BROW + kp];
    #pragma unroll
    for (int c = 0; c < 8; c++) {
        float4 f = ((const float4*)gp)[c];
        sp[c * 4 + 0] = (__bf16)f.x;
        sp[c * 4 + 1] = (__bf16)f.y;
        sp[c * 4 + 2] = (__bf16)f.z;
        sp[c * 4 + 3] = (__bf16)f.w;
    }
    if (prefetch_next) __builtin_prefetch(gp + BK, 0, 1);
}

// ---------------------------------------------------------------------------
// Kernel 4: FFN1 per expert: h = gelu(z[tok] @ w1[e]^T + b1[e])  -> bf16
// ---------------------------------------------------------------------------
__global__ __launch_bounds__(256)
void moe_ffn1_kernel(const float* __restrict__ z, const float* __restrict__ w1,
                     const float* __restrict__ b1, const int* __restrict__ segOff,
                     const int* __restrict__ assign, unsigned short* __restrict__ hraw) {
    const int e    = blockIdx.x;
    const int base = segOff[e];
    const int cnt  = segOff[e + 1] - base;
    const int mTile = blockIdx.y * BM;
    if (mTile >= cnt) return;                        // uniform early-exit
    const int nTile = blockIdx.z * BN;

    __shared__ __bf16 Bs[BN * BROW];                 // 18.4 KB
    __bf16* hbuf = (__bf16*)hraw;

    const int tid = threadIdx.x;
    const int lane = tid & 31, wid = tid >> 5;
    const int wm = wid & 3, wn = wid >> 2;           // 4 x 2 wave grid
    const int half = lane >> 4, l15 = lane & 15;

    const int mA = mTile + wm * 16 + l15;            // A row owned by this lane
    const float* zrow = nullptr;
    if (mA < cnt) zrow = z + (size_t)(assign[base + mA] >> 1) * DIM_D;

    const float* w1e = w1 + (size_t)e * DIM_H * DIM_D + (size_t)nTile * DIM_D;

    v8f acc0 = {0,0,0,0,0,0,0,0};
    v8f acc1 = {0,0,0,0,0,0,0,0};
    v8f acc2 = {0,0,0,0,0,0,0,0};
    v8f acc3 = {0,0,0,0,0,0,0,0};

    for (int k0 = 0; k0 < DIM_D; k0 += BK) {
        load_b_tile(w1e, DIM_D, k0, Bs, tid, k0 + BK < DIM_D);
        __syncthreads();

        v16bf a0, a1;
        if (zrow) {
            a0 = load_a_f32(zrow + k0 + half * 8);
            a1 = load_a_f32(zrow + k0 + 32 + half * 8);
        } else {
            a0 = zero_a(); a1 = zero_a();
        }

        const int nb = wn * 64 + l15;
        #pragma unroll
        for (int ks = 0; ks < 2; ks++) {
            const v16bf a = ks ? a1 : a0;
            const int kb = half * 8 + ks * 32;
            v16bf b0 = load_b_frag(Bs, nb,      kb);
            v16bf b1f = load_b_frag(Bs, nb + 16, kb);
            v16bf b2f = load_b_frag(Bs, nb + 32, kb);
            v16bf b3f = load_b_frag(Bs, nb + 48, kb);
            acc0 = __builtin_amdgcn_wmma_f32_16x16x32_bf16(false, a, false, b0,  (short)0, acc0, false, false);
            acc1 = __builtin_amdgcn_wmma_f32_16x16x32_bf16(false, a, false, b1f, (short)0, acc1, false, false);
            acc2 = __builtin_amdgcn_wmma_f32_16x16x32_bf16(false, a, false, b2f, (short)0, acc2, false, false);
            acc3 = __builtin_amdgcn_wmma_f32_16x16x32_bf16(false, a, false, b3f, (short)0, acc3, false, false);
        }
        __syncthreads();
    }

    // epilogue: bias + exact GELU, store bf16 h
    const v8f accs[4] = {acc0, acc1, acc2, acc3};
    #pragma unroll
    for (int f = 0; f < 4; f++) {
        const int nG = nTile + wn * 64 + f * 16 + l15;
        const float bias = b1[e * DIM_H + nG];
        #pragma unroll
        for (int r = 0; r < 8; r++) {
            const int m = mTile + wm * 16 + r + 8 * half;
            if (m < cnt) {
                float x = accs[f][r] + bias;
                float g = 0.5f * x * (1.0f + erff(x * 0.70710678f));
                hbuf[(size_t)(base + m) * DIM_H + nG] = (__bf16)g;
            }
        }
    }
}

// ---------------------------------------------------------------------------
// Kernel 5: FFN2 per expert: out[tok] += w * (h @ w2[e]^T + b2[e])
// A comes pre-converted (bf16) from hbuf; scatter via atomicAdd (exactly two
// commutative adds per output element -> deterministic).
// ---------------------------------------------------------------------------
__global__ __launch_bounds__(256)
void moe_ffn2_kernel(const unsigned short* __restrict__ hraw, const float* __restrict__ w2,
                     const float* __restrict__ b2, const int* __restrict__ segOff,
                     const int* __restrict__ assign, const float* __restrict__ topk_w,
                     float* __restrict__ out) {
    const int e    = blockIdx.x;
    const int base = segOff[e];
    const int cnt  = segOff[e + 1] - base;
    const int mTile = blockIdx.y * BM;
    if (mTile >= cnt) return;
    const int nTile = blockIdx.z * BN;

    __shared__ __bf16 Bs[BN * BROW];
    const __bf16* hbuf = (const __bf16*)hraw;

    const int tid = threadIdx.x;
    const int lane = tid & 31, wid = tid >> 5;
    const int wm = wid & 3, wn = wid >> 2;
    const int half = lane >> 4, l15 = lane & 15;

    const int mA = mTile + wm * 16 + l15;
    const __bf16* hrow = nullptr;
    if (mA < cnt) hrow = hbuf + (size_t)(base + mA) * DIM_H;

    const float* w2e = w2 + (size_t)e * DIM_O * DIM_H + (size_t)nTile * DIM_H;

    v8f acc0 = {0,0,0,0,0,0,0,0};
    v8f acc1 = {0,0,0,0,0,0,0,0};
    v8f acc2 = {0,0,0,0,0,0,0,0};
    v8f acc3 = {0,0,0,0,0,0,0,0};

    for (int k0 = 0; k0 < DIM_H; k0 += BK) {
        load_b_tile(w2e, DIM_H, k0, Bs, tid, k0 + BK < DIM_H);
        __syncthreads();

        v16bf a0, a1;
        if (hrow) {
            const __bf16* ap = hrow + k0 + half * 8;       // all 16B-aligned
            uint4 u0 = ((const uint4*)ap)[0];              // K: +0..7
            uint4 u1 = ((const uint4*)(ap + 16))[0];       // K: +16..23
            uint4 u2 = ((const uint4*)(ap + 32))[0];       // K: +32..39
            uint4 u3 = ((const uint4*)(ap + 48))[0];       // K: +48..55
            a0 = bitcast_v16bf(u0, u1);
            a1 = bitcast_v16bf(u2, u3);
        } else {
            a0 = zero_a(); a1 = zero_a();
        }

        const int nb = wn * 64 + l15;
        #pragma unroll
        for (int ks = 0; ks < 2; ks++) {
            const v16bf a = ks ? a1 : a0;
            const int kb = half * 8 + ks * 32;
            v16bf b0 = load_b_frag(Bs, nb,      kb);
            v16bf b1f = load_b_frag(Bs, nb + 16, kb);
            v16bf b2f = load_b_frag(Bs, nb + 32, kb);
            v16bf b3f = load_b_frag(Bs, nb + 48, kb);
            acc0 = __builtin_amdgcn_wmma_f32_16x16x32_bf16(false, a, false, b0,  (short)0, acc0, false, false);
            acc1 = __builtin_amdgcn_wmma_f32_16x16x32_bf16(false, a, false, b1f, (short)0, acc1, false, false);
            acc2 = __builtin_amdgcn_wmma_f32_16x16x32_bf16(false, a, false, b2f, (short)0, acc2, false, false);
            acc3 = __builtin_amdgcn_wmma_f32_16x16x32_bf16(false, a, false, b3f, (short)0, acc3, false, false);
        }
        __syncthreads();
    }

    const v8f accs[4] = {acc0, acc1, acc2, acc3};
    #pragma unroll
    for (int f = 0; f < 4; f++) {
        const int nG = nTile + wn * 64 + f * 16 + l15;
        const float bias = b2[e * DIM_O + nG];
        #pragma unroll
        for (int r = 0; r < 8; r++) {
            const int m = mTile + wm * 16 + r + 8 * half;
            if (m < cnt) {
                const int av = assign[base + m];          // token*2 + slot
                const int token = av >> 1;
                const float wgt = topk_w[av];
                atomicAdd(&out[(size_t)token * DIM_O + nG], wgt * (accs[f][r] + bias));
            }
        }
    }
}

// ---------------------------------------------------------------------------
// Host launcher
// ---------------------------------------------------------------------------
extern "C" void kernel_launch(void* const* d_in, const int* in_sizes, int n_in,
                              void* d_out, int out_size, void* d_ws, size_t ws_size,
                              hipStream_t stream) {
    const float* z  = (const float*)d_in[0];
    const float* rw = (const float*)d_in[1];
    const float* rb = (const float*)d_in[2];
    const float* w1 = (const float*)d_in[3];
    const float* b1 = (const float*)d_in[4];
    const float* w2 = (const float*)d_in[5];
    const float* b2 = (const float*)d_in[6];
    float* out = (float*)d_out;

    // workspace carve (256B aligned)
    char* ws = (char*)d_ws;
    auto carve = [&](size_t bytes) -> char* {
        char* p = ws;
        ws += (bytes + 255) & ~(size_t)255;
        return p;
    };
    int*            topk_idx = (int*)  carve((size_t)N_TOK * TOPK * 4);
    float*          topk_w   = (float*)carve((size_t)N_TOK * TOPK * 4);
    int*            counts   = (int*)  carve(N_EXP * 4);
    int*            segOff   = (int*)  carve((N_EXP + 1) * 4);
    int*            cursor   = (int*)  carve(N_EXP * 4);
    int*            assign   = (int*)  carve((size_t)N_TOK * TOPK * 4);
    unsigned short* hbuf     = (unsigned short*)carve((size_t)N_TOK * TOPK * DIM_H * 2); // 16 MB bf16

    const int totalOut = N_TOK * DIM_O;
    moe_zero_kernel<<<(totalOut + 255) / 256, 256, 0, stream>>>(out, totalOut, counts);
    moe_router_kernel<<<N_TOK / 8, 256, 0, stream>>>(z, rw, rb, topk_idx, topk_w, counts);
    moe_scan_kernel<<<1, 32, 0, stream>>>(counts, segOff, cursor);
    moe_place_kernel<<<(N_TOK * TOPK + 255) / 256, 256, 0, stream>>>(topk_idx, cursor, assign);

    dim3 gB(N_EXP, N_TOK / BM, DIM_H / BN);   // (16, 32, 16): worst-case M coverage
    moe_ffn1_kernel<<<gB, 256, 0, stream>>>(z, w1, b1, segOff, assign, hbuf);

    dim3 gC(N_EXP, N_TOK / BM, DIM_O / BN);   // (16, 32, 8)
    moe_ffn2_kernel<<<gC, 256, 0, stream>>>(hbuf, w2, b2, segOff, assign, topk_w, out);
}
